// TemporalExt_47991964566192
// MI455X (gfx1250) — matrix operand order repaired and checked by reading, not compiled, not run on hardware
//
#include <hip/hip_runtime.h>
#include <math.h>

// ---- problem dims (fixed by reference) ----
#define B_   8
#define L_   256
#define F_   4
#define D_   256     // d_model
#define ED_  512     // d_inner
#define N_   16      // d_state
#define DT_  16      // dt_rank
#define NL_  2
#define V_   128
#define K_   4
#define OUT_ 256

typedef __attribute__((ext_vector_type(16))) __bf16 v16bf;
typedef __attribute__((ext_vector_type(8)))  float  v8f;

__device__ __forceinline__ float siluf(float v) {
    return v / (1.f + __expf(-v));
}
__device__ __forceinline__ float softplusf(float v) {
    return (v > 20.f) ? v : log1pf(__expf(v));
}

// ---------------------------------------------------------------------------
// gather: x (B,L,F,C,E) -> h (F,B,L,D)   (D = C*E contiguous in source)
// ---------------------------------------------------------------------------
__global__ void gather_kernel(const float* __restrict__ x, float* __restrict__ h) {
    long idx = (long)blockIdx.x * 256 + threadIdx.x;
    if (idx >= (long)F_ * B_ * L_ * D_) return;
    int d = (int)(idx & (D_ - 1));
    long row = idx >> 8;           // / D_
    int l = (int)(row % L_);
    long fb = row / L_;
    int b = (int)(fb % B_);
    int f = (int)(fb / B_);
    long src = (((long)b * L_ + l) * F_ + f) * D_ + d;
    h[idx] = x[src];
}

// ---------------------------------------------------------------------------
// RMSNorm: one 256-thread block per row (F*B*L rows), D=256 elems
// ---------------------------------------------------------------------------
__global__ void rmsnorm_kernel(const float* __restrict__ in, float* __restrict__ out,
                               const float* __restrict__ w, int wFeatStride) {
    __shared__ float red[256];
    int row = blockIdx.x;
    int f = row / (B_ * L_);
    int tid = threadIdx.x;
    float xv = in[(long)row * D_ + tid];
    red[tid] = xv * xv;
    __syncthreads();
    for (int s = 128; s > 0; s >>= 1) {
        if (tid < s) red[tid] += red[tid + s];
        __syncthreads();
    }
    float rms = rsqrtf(red[0] * (1.f / D_) + 1e-5f);
    out[(long)row * D_ + tid] = xv * rms * w[(long)f * wFeatStride + tid];
}

// ---------------------------------------------------------------------------
// Weight packer: W (per-feature, N x K f32, row stride ldw) -> bf16 in WMMA
// B-fragment order:
//   out[(((f*Ntiles + nt)*KSTEPS + ks)*32 + lane)*16 + e]
//     = W[f][nt*16 + (lane&15)][ks*32 + 8*(lane>>4) + (e&7) + 16*(e>>3)]   (0 beyond KACT)
// ---------------------------------------------------------------------------
template <int KACT>
__global__ void pack_w_kernel(const float* __restrict__ W, int wFeatStride, int ldw,
                              __bf16* __restrict__ outp, int Ntiles, long total) {
    constexpr int KPAD   = (KACT + 31) & ~31;
    constexpr int KSTEPS = KPAD / 32;
    long idx = (long)blockIdx.x * 256 + threadIdx.x;
    if (idx >= total) return;
    int e    = (int)(idx & 15);
    int lane = (int)((idx >> 4) & 31);
    long t   = idx >> 9;
    int ks   = (int)(t % KSTEPS);
    t /= KSTEPS;
    int nt   = (int)(t % Ntiles);
    int f    = (int)(t / Ntiles);
    int n = nt * 16 + (lane & 15);
    int k = ks * 32 + 8 * (lane >> 4) + (e & 7) + 16 * (e >> 3);
    float v = (k < KACT) ? W[(long)f * wFeatStride + (long)n * ldw + k] : 0.f;
    outp[idx] = (__bf16)v;
}

// ---------------------------------------------------------------------------
// WMMA bf16 GEMM:  Out[M,N] = act[M,KACT] . W[N,KACT]^T  (+ epilogue)
// One wave computes 1 M-tile x NT N-tiles (NT independent accumulators so
// back-to-back WMMAs fill each other's D->C hazard slots).
// grid = (M/16, Ntiles/NT, F). K fully unrolled.
// A fragment (ISA 7.12.2): lane half=lane>>4, m=lane&15; elems 0..7 are
//   k = k0+8*half+0..7, elems 8..15 are k = k0+16+8*half+0..7  -> 4x b128.
// B fragments come pre-packed in bf16 (one aligned v16bf load per tile/step).
// C/D: vgpr r -> m = r + 8*half; n = lane&15.
// mode: 0 plain | 1 bias+softplus | 2 accumulate | 3 bias+permuted store
// ---------------------------------------------------------------------------
template <int KACT, int NT>
__global__ void __launch_bounds__(32)
gemm_wmma_kernel(const float* __restrict__ act, int actFeatStride, int lda,
                 const __bf16* __restrict__ Bp,
                 const float* __restrict__ bias, int biasFeatStride,
                 float* out, int outFeatStride, int ldo,
                 int mode) {
    constexpr int KPAD   = (KACT + 31) & ~31;
    constexpr int KSTEPS = KPAD / 32;
    const int f      = blockIdx.z;
    const int m0     = blockIdx.x * 16;
    const int ntBase = blockIdx.y * NT;
    const int ntTot  = gridDim.y * NT;
    const int lane   = threadIdx.x;
    const int half   = lane >> 4;
    const int l16    = lane & 15;

    const float* Ar = act + (long)f * actFeatStride + (long)(m0 + l16) * lda;
    const __bf16* Bl = Bp + ((((long)f * ntTot + ntBase) * KSTEPS) * 32 + lane) * 16;

    __builtin_prefetch(Ar, 0, 1);
    __builtin_prefetch(Bl, 0, 1);

    const v8f vzero = {0.f, 0.f, 0.f, 0.f, 0.f, 0.f, 0.f, 0.f};
    v8f acc[NT];
    #pragma unroll
    for (int j = 0; j < NT; ++j) acc[j] = vzero;

    #pragma unroll
    for (int ks = 0; ks < KSTEPS; ++ks) {
        const int k0 = ks * 32;
        v16bf afrag;
        #pragma unroll
        for (int g = 0; g < 2; ++g) {
            const int kgb = k0 + g * 16;           // compile-time
            if (kgb + 16 <= KACT) {                // compile-time: whole group valid
                const float4* p = (const float4*)(Ar + kgb + half * 8);
                float4 lo = p[0];
                float4 hi = p[1];
                afrag[g * 8 + 0] = (__bf16)lo.x;
                afrag[g * 8 + 1] = (__bf16)lo.y;
                afrag[g * 8 + 2] = (__bf16)lo.z;
                afrag[g * 8 + 3] = (__bf16)lo.w;
                afrag[g * 8 + 4] = (__bf16)hi.x;
                afrag[g * 8 + 5] = (__bf16)hi.y;
                afrag[g * 8 + 6] = (__bf16)hi.z;
                afrag[g * 8 + 7] = (__bf16)hi.w;
            } else {                               // compile-time: whole group is pad
                #pragma unroll
                for (int j = 0; j < 8; ++j) afrag[g * 8 + j] = (__bf16)0.f;
            }
        }
        v16bf bfrag[NT];
        #pragma unroll
        for (int j = 0; j < NT; ++j)
            bfrag[j] = *(const v16bf*)(Bl + ((long)j * KSTEPS + ks) * 512);
        #pragma unroll
        for (int j = 0; j < NT; ++j)
            acc[j] = __builtin_amdgcn_wmma_f32_16x16x32_bf16(
                false, afrag, false, bfrag[j], (short)0, acc[j], false, false);
    }

    #pragma unroll
    for (int j = 0; j < NT; ++j) {
        const int n0 = (ntBase + j) * 16;
        const int n  = n0 + l16;
        #pragma unroll
        for (int r = 0; r < 8; ++r) {
            int m = m0 + r + 8 * half;
            float v = acc[j][r];
            if (mode == 0) {
                out[(long)f * outFeatStride + (long)m * ldo + n] = v;
            } else if (mode == 1) {
                v += bias[(long)f * biasFeatStride + n];
                out[(long)f * outFeatStride + (long)m * ldo + n] = softplusf(v);
            } else if (mode == 2) {
                out[(long)f * outFeatStride + (long)m * ldo + n] += v;
            } else {
                // rows are (F,B,L)-ordered; store to (B,F,L) order
                v += bias[n];
                int ff = m / (B_ * L_);
                int rr = m - ff * (B_ * L_);
                int bb = rr / L_;
                int ll = rr - bb * L_;
                long orow = ((long)bb * F_ + ff) * L_ + ll;
                out[orow * (long)ldo + n] = v;
            }
        }
    }
}

// ---------------------------------------------------------------------------
// Causal depthwise conv1d (K=4) + bias + SiLU. One thread per (f,b,ch).
// ---------------------------------------------------------------------------
__global__ void conv_silu_kernel(const float* __restrict__ xz, float* __restrict__ xcs,
                                 const float* __restrict__ cw, const float* __restrict__ cb,
                                 int layer) {
    int t = blockIdx.x * 256 + threadIdx.x;
    if (t >= F_ * B_ * ED_) return;
    int f  = t / (B_ * ED_);
    int r  = t - f * (B_ * ED_);
    int b  = r / ED_;
    int ch = r - b * ED_;

    const float* w = cw + (((long)f * NL_ + layer) * ED_ + ch) * K_;
    float w0 = w[0], w1 = w[1], w2 = w[2], w3 = w[3];
    float bias = cb[((long)f * NL_ + layer) * ED_ + ch];

    const float* src = xz  + ((long)(f * B_ + b)) * L_ * (2 * ED_) + ch;
    float*       dst = xcs + ((long)(f * B_ + b)) * L_ * ED_ + ch;

    float x0 = 0.f, x1 = 0.f, x2 = 0.f;
    for (int li = 0; li < L_; ++li) {
        float x3 = src[(long)li * (2 * ED_)];
        float v = w0 * x0 + w1 * x1 + w2 * x2 + w3 * x3 + bias;
        dst[(long)li * ED_] = siluf(v);
        x0 = x1; x1 = x2; x2 = x3;
    }
}

// ---------------------------------------------------------------------------
// Selective-scan + gating. One thread per (f,b,ch); 16 states in registers.
// Reads delta from dy, overwrites dy with gated y. B/C rows via float4.
// ---------------------------------------------------------------------------
__global__ void scan_kernel(float* dy,
                            const float* __restrict__ xcs,
                            const float* __restrict__ xz,
                            const float* __restrict__ xdbl,
                            const float* __restrict__ A_log,
                            const float* __restrict__ D_p,
                            int layer) {
    int t = blockIdx.x * 256 + threadIdx.x;
    if (t >= F_ * B_ * ED_) return;
    int f  = t / (B_ * ED_);
    int r  = t - f * (B_ * ED_);
    int b  = r / ED_;
    int ch = r - b * ED_;

    const float* ar = A_log + (((long)f * NL_ + layer) * ED_ + ch) * N_;
    float An[N_];
    #pragma unroll
    for (int n = 0; n < N_; ++n) An[n] = -__expf(ar[n]);
    float dpv = D_p[((long)f * NL_ + layer) * ED_ + ch];

    float hs[N_];
    #pragma unroll
    for (int n = 0; n < N_; ++n) hs[n] = 0.f;

    long rowbase = (long)(f * B_ + b) * L_;
    for (int li = 0; li < L_; ++li) {
        long row = rowbase + li;
        float dv = dy[row * ED_ + ch];
        float xv = xcs[row * ED_ + ch];
        const float4* bc4 = (const float4*)(xdbl + row * (DT_ + 2 * N_) + DT_);
        float4 Bq[4], Cq[4];
        #pragma unroll
        for (int q = 0; q < 4; ++q) Bq[q] = bc4[q];
        #pragma unroll
        for (int q = 0; q < 4; ++q) Cq[q] = bc4[4 + q];
        const float* Bv = (const float*)Bq;
        const float* Cv = (const float*)Cq;
        float y = 0.f;
        float dx = dv * xv;
        #pragma unroll
        for (int n = 0; n < N_; ++n) {
            float da = __expf(dv * An[n]);
            hs[n] = da * hs[n] + dx * Bv[n];
            y += hs[n] * Cv[n];
        }
        y += dpv * xv;
        float zv = xz[row * (2 * ED_) + ED_ + ch];
        y *= siluf(zv);
        dy[row * ED_ + ch] = y;
    }
}

// ---------------------------------------------------------------------------
// BatchNorm: per-feature mean/var over (B,L,V), then normalize in place.
// ---------------------------------------------------------------------------
__global__ void bnstats_kernel(const float* __restrict__ logits, float* __restrict__ stats) {
    __shared__ float s1[256], s2[256];
    int f = blockIdx.x, tid = threadIdx.x;
    const int NTOT = B_ * L_ * V_;
    const float* base = logits + (long)f * NTOT;
    float a = 0.f, b2 = 0.f;
    for (int i = tid; i < NTOT; i += 256) { float v = base[i]; a += v; b2 += v * v; }
    s1[tid] = a; s2[tid] = b2;
    __syncthreads();
    for (int s = 128; s > 0; s >>= 1) {
        if (tid < s) { s1[tid] += s1[tid + s]; s2[tid] += s2[tid + s]; }
        __syncthreads();
    }
    if (tid == 0) {
        float n  = (float)NTOT;
        float mu = s1[0] / n;
        float var = s2[0] / n - mu * mu;
        stats[2 * f]     = mu;
        stats[2 * f + 1] = rsqrtf(var + 1e-5f);
    }
}

__global__ void bnapply_kernel(float* logits, const float* __restrict__ stats,
                               const float* __restrict__ gamma, const float* __restrict__ beta) {
    long idx = (long)blockIdx.x * 256 + threadIdx.x;
    if (idx >= (long)F_ * B_ * L_ * V_) return;
    int f = (int)(idx / (B_ * L_ * V_));
    float v = logits[idx];
    logits[idx] = (v - stats[2 * f]) * stats[2 * f + 1] * gamma[f] + beta[f];
}

// ---------------------------------------------------------------------------
// host orchestration
// ---------------------------------------------------------------------------
extern "C" void kernel_launch(void* const* d_in, const int* in_sizes, int n_in,
                              void* d_out, int out_size, void* d_ws, size_t ws_size,
                              hipStream_t stream) {
    (void)in_sizes; (void)n_in; (void)out_size; (void)ws_size;
    const float* x    = (const float*)d_in[0];
    const float* ipw  = (const float*)d_in[1];
    const float* cw   = (const float*)d_in[2];
    const float* cb   = (const float*)d_in[3];
    const float* xpw  = (const float*)d_in[4];
    const float* dtw  = (const float*)d_in[5];
    const float* dtb  = (const float*)d_in[6];
    const float* alog = (const float*)d_in[7];
    const float* dp   = (const float*)d_in[8];
    const float* opw  = (const float*)d_in[9];
    const float* nw   = (const float*)d_in[10];
    const float* fnw  = (const float*)d_in[11];
    const float* lhw  = (const float*)d_in[12];
    const float* gam  = (const float*)d_in[13];
    const float* bet  = (const float*)d_in[14];
    const float* tw   = (const float*)d_in[15];
    const float* tb   = (const float*)d_in[16];
    float* out = (float*)d_out;
    float* ws  = (float*)d_ws;

    // workspace layout (floats)
    float* h      = ws;                    // (F,B,L,D)          2,097,152
    float* hn     = h      + 2097152;      // (F,B,L,D)          2,097,152
    float* xz     = hn     + 2097152;      // (F,B,L,2ED)        8,388,608
    float* xcs    = xz     + 8388608;      // (F,B,L,ED)         4,194,304
    float* xdbl   = xcs    + 4194304;      // (F,B,L,48)           393,216
    float* dy     = xdbl   + 393216;       // (F,B,L,ED) delta->y 4,194,304
    float* logits = dy     + 4194304;      // (F,B,L,V)          1,048,576
    float* stats  = logits + 1048576;      // (F,2) + pad            8
    __bf16* pb    = (__bf16*)(stats + 8);  // packed weights (<= 2 MB)

    // x -> residual stream h
    gather_kernel<<<(F_ * B_ * L_ * D_) / 256, 256, 0, stream>>>(x, h);

    for (int l = 0; l < NL_; ++l) {
        // pre-norm
        rmsnorm_kernel<<<F_ * B_ * L_, 256, 0, stream>>>(h, hn, nw + (long)l * D_, NL_ * D_);

        // in_proj: (2048,256) x (1024,256)^T -> xz   [K=256, Ntiles=64, NT=4]
        {
            long total = (long)F_ * 64 * 8 * 512;
            pack_w_kernel<256><<<(int)((total + 255) / 256), 256, 0, stream>>>(
                ipw + (long)l * (2 * ED_) * D_, NL_ * 2 * ED_ * D_, D_, pb, 64, total);
            gemm_wmma_kernel<256, 4><<<dim3(128, 16, F_), 32, 0, stream>>>(
                hn, B_ * L_ * D_, D_, pb, nullptr, 0,
                xz, B_ * L_ * 2 * ED_, 2 * ED_, 0);
        }

        // depthwise conv + silu
        conv_silu_kernel<<<(F_ * B_ * ED_ + 255) / 256, 256, 0, stream>>>(xz, xcs, cw, cb, l);

        // x_proj: (2048,512) x (48,512)^T -> xdbl   [K=512, Ntiles=3, NT=3]
        {
            long total = (long)F_ * 3 * 16 * 512;
            pack_w_kernel<512><<<(int)((total + 255) / 256), 256, 0, stream>>>(
                xpw + (long)l * (DT_ + 2 * N_) * ED_, NL_ * (DT_ + 2 * N_) * ED_, ED_,
                pb, 3, total);
            gemm_wmma_kernel<512, 3><<<dim3(128, 1, F_), 32, 0, stream>>>(
                xcs, B_ * L_ * ED_, ED_, pb, nullptr, 0,
                xdbl, B_ * L_ * (DT_ + 2 * N_), DT_ + 2 * N_, 0);
        }

        // dt_proj + bias + softplus: (2048,16 pad 32) x (512,16)^T -> delta
        {
            long total = (long)F_ * 32 * 1 * 512;
            pack_w_kernel<16><<<(int)((total + 255) / 256), 256, 0, stream>>>(
                dtw + (long)l * ED_ * DT_, NL_ * ED_ * DT_, DT_, pb, 32, total);
            gemm_wmma_kernel<16, 4><<<dim3(128, 8, F_), 32, 0, stream>>>(
                xdbl, B_ * L_ * (DT_ + 2 * N_), DT_ + 2 * N_, pb,
                dtb + (long)l * ED_, NL_ * ED_,
                dy, B_ * L_ * ED_, ED_, 1);
        }

        // selective scan + SiLU(z) gating (writes y into dy)
        scan_kernel<<<(F_ * B_ * ED_ + 255) / 256, 256, 0, stream>>>(
            dy, xcs, xz, xdbl, alog, dp, l);

        // out_proj with residual accumulate into h   [K=512, Ntiles=16, NT=4]
        {
            long total = (long)F_ * 16 * 16 * 512;
            pack_w_kernel<512><<<(int)((total + 255) / 256), 256, 0, stream>>>(
                opw + (long)l * D_ * ED_, NL_ * D_ * ED_, ED_, pb, 16, total);
            gemm_wmma_kernel<512, 4><<<dim3(128, 4, F_), 32, 0, stream>>>(
                dy, B_ * L_ * ED_, ED_, pb, nullptr, 0,
                h, B_ * L_ * D_, D_, 2);
        }
    }

    // final norm + lm_head   [K=256, Ntiles=8, NT=4]
    rmsnorm_kernel<<<F_ * B_ * L_, 256, 0, stream>>>(h, hn, fnw, D_);
    {
        long total = (long)F_ * 8 * 8 * 512;
        pack_w_kernel<256><<<(int)((total + 255) / 256), 256, 0, stream>>>(
            lhw, V_ * D_, D_, pb, 8, total);
        gemm_wmma_kernel<256, 4><<<dim3(128, 2, F_), 32, 0, stream>>>(
            hn, B_ * L_ * D_, D_, pb, nullptr, 0,
            logits, B_ * L_ * V_, V_, 0);
    }

    // BatchNorm (training-mode batch stats) in place
    bnstats_kernel<<<F_, 256, 0, stream>>>(logits, stats);
    bnapply_kernel<<<(F_ * B_ * L_ * V_) / 256, 256, 0, stream>>>(logits, stats, gam, bet);

    // temb: (F*B*L, 128) x (256,128)^T + b, permuted store to (B,F,L,256)
    {
        long total = (long)1 * 16 * 4 * 512;   // single "feature"
        pack_w_kernel<128><<<(int)((total + 255) / 256), 256, 0, stream>>>(
            tw, 0, V_, pb, 16, total);
        gemm_wmma_kernel<128, 4><<<dim3((F_ * B_ * L_) / 16, 4, 1), 32, 0, stream>>>(
            logits, 0, V_, pb, tb, 0,
            out, 0, OUT_, 3);
    }
}
